// SAKEInteraction_90022514524667
// MI455X (gfx1250) — matrix operand order repaired and checked by reading, not compiled, not run on hardware
//
#include <hip/hip_runtime.h>

#define NA 10000
#define NP 160000

typedef __attribute__((ext_vector_type(16))) _Float16 v16h;
typedef __attribute__((ext_vector_type(8)))  float    v8f;

__device__ __forceinline__ float silu_f(float z) { return z / (1.f + __expf(-z)); }

// ---- WMMA fragment-major LDS layouts (wave32, v_wmma_f32_16x16x32_f16, ISA 7.12.2) ----
// Each 16x32 (A) or 32x16 (B) f16 tile is stored as tile[lane][16 halves] = 512 halves,
// so a fragment load is one contiguous 32B read per lane (2x ds_load_b128).

// inverse map for A tiles: element (m, k) of a 16x32 tile -> half index
__device__ __forceinline__ int a_slot(int m, int k) {
  return (((m & 15) + ((k & 8) ? 16 : 0)) << 4) + ((k & 16) ? 8 : 0) + (k & 7);
}
// multi-K-tile A matrix (tiles of 512 halves, K-tile major)
__device__ __forceinline__ int a_idx(int m, int k) {
  return ((k >> 5) << 9) + a_slot(m, k & 31);
}
// inverse map for B tiles: element (k, n) of a 32x16 tile -> half index
__device__ __forceinline__ int b_slot(int k, int n) {
  return (((n & 15) + ((k & 8) ? 16 : 0)) << 4) + ((k & 16) ? 8 : 0) + (k & 7);
}
// B matrix with ntn N-tiles: tile order (ktile * ntn + ntile)
__device__ __forceinline__ int b_idx(int k, int n, int ntn) {
  return (((k >> 5) * ntn + (n >> 4)) << 9) + b_slot(k & 31, n & 15);
}

__device__ __forceinline__ v16h frag_ld(const v16h* tile, int lane) { return tile[lane]; }

__device__ __forceinline__ v8f wmma32(v16h a, v16h b, v8f c) {
  return __builtin_amdgcn_wmma_f32_16x16x32_f16(false, a, false, b, (short)0, c, false, false);
}

// ordered-uint transform for float atomicMax (monotone)
__device__ __forceinline__ unsigned f2ord(float f) {
  unsigned u = __float_as_uint(f);
  return (u & 0x80000000u) ? ~u : (u | 0x80000000u);
}
__device__ __forceinline__ float ord2f(unsigned u) {
  return (u & 0x80000000u) ? __uint_as_float(u & 0x7fffffffu) : __uint_as_float(~u);
}

// ---------------------------------------------------------------------------
__global__ void k_init(unsigned* base, long long n) {
  long long i = (long long)blockIdx.x * blockDim.x + threadIdx.x;
  long long st = (long long)gridDim.x * blockDim.x;
  for (; i < n; i += st) base[i] = 0u;
}

// ---------------------------------------------------------------------------
// Edge model: 16 pairs per wave, 4 waves per block, all dense math via WMMA f16.
__global__ void __launch_bounds__(128) k_edge(
    const float* __restrict__ h, const float* __restrict__ x, const int* __restrict__ pair,
    const float* __restrict__ w_mlp_in, const float* __restrict__ b_mlp_in,
    const float* __restrict__ w_e1, const float* __restrict__ b_e1,
    const float* __restrict__ w_e2, const float* __restrict__ b_e2,
    const float* __restrict__ w_att, const float* __restrict__ b_att,
    float* __restrict__ ws_d, float* __restrict__ ws_dir,
    float* __restrict__ ws_edge, float* __restrict__ ws_log, float* __restrict__ counts) {
  __shared__ v16h sWm[8 * 32];          // w_mlp_in: 2 K-tiles x 4 N-tiles (cols padded 50->64)
  __shared__ v16h sW1[8 * 32];          // w_e1: 4 K-tiles x 2 N-tiles (rows padded 115->128)
  __shared__ v16h sW2[2 * 32];          // w_e2: 1 K x 2 N
  __shared__ v16h sWa[1 * 32];          // w_att: 1 K x 1 N (cols padded 7->16)
  __shared__ float sBm[64], sB1[32], sB2[32], sBa[16];
  __shared__ v16h sHc[4][2 * 32];       // A: 16x64
  __shared__ _Float16 sG1[4][16 * 64];  // linear (elementwise only)
  __shared__ v16h sEin[4][4 * 32];      // A: 16x128
  __shared__ v16h sMid[4][32];          // A: 16x32
  __shared__ v16h sEdg[4][32];          // A: 16x32
  __shared__ float sDv[4][16];
  __shared__ int sSi[4][16], sSj[4][16];

  int tid = threadIdx.x;
  {
    _Float16* wm = (_Float16*)sWm;
    _Float16* w1 = (_Float16*)sW1;
    _Float16* w2 = (_Float16*)sW2;
    _Float16* wa = (_Float16*)sWa;
    for (int t = tid; t < 64 * 64; t += 128) {
      int k = t >> 6, c = t & 63;
      wm[b_idx(k, c, 4)] = (_Float16)((c < 50) ? w_mlp_in[k * 50 + c] : 0.f);
    }
    for (int t = tid; t < 128 * 32; t += 128) {
      int k = t >> 5, c = t & 31;
      w1[b_idx(k, c, 2)] = (_Float16)((k < 115) ? w_e1[t] : 0.f);
    }
    for (int t = tid; t < 32 * 32; t += 128) {
      int k = t >> 5, c = t & 31;
      w2[b_idx(k, c, 2)] = (_Float16)w_e2[t];
    }
    for (int t = tid; t < 32 * 16; t += 128) {
      int k = t >> 4, c = t & 15;
      wa[b_idx(k, c, 1)] = (_Float16)((c < 7) ? w_att[k * 7 + c] : 0.f);
    }
  }
  if (tid < 64) sBm[tid] = (tid < 50) ? b_mlp_in[tid] : 0.f;
  if (tid < 32) { sB1[tid] = b_e1[tid]; sB2[tid] = b_e2[tid]; }
  if (tid < 16) sBa[tid] = (tid < 7) ? b_att[tid] : 0.f;
  __syncthreads();

  int wave = tid >> 5, lane = tid & 31;
  int p0 = (blockIdx.x * 4 + wave) * 16;
  _Float16* hcw  = (_Float16*)&sHc[wave][0];
  _Float16* einw = (_Float16*)&sEin[wave][0];
  _Float16* midw = (_Float16*)&sMid[wave][0];
  _Float16* edgw = (_Float16*)&sEdg[wave][0];

  if (lane < 16) {
    int p = p0 + lane;
    int i = pair[p], j = pair[NP + p];
    sSi[wave][lane] = i; sSj[wave][lane] = j;
    float rx = x[j * 3 + 0] - x[i * 3 + 0];
    float ry = x[j * 3 + 1] - x[i * 3 + 1];
    float rz = x[j * 3 + 2] - x[i * 3 + 2];
    float d = sqrtf(rx * rx + ry * ry + rz * rz);
    float inv = 1.f / (d + 1e-5f);
    sDv[wave][lane] = d;
    ws_d[p] = d;
    ws_dir[p * 3 + 0] = rx * inv; ws_dir[p * 3 + 1] = ry * inv; ws_dir[p * 3 + 2] = rz * inv;
    atomicAdd(&counts[i], 1.f);
  }
  __syncthreads();

  for (int t = lane; t < 16 * 64; t += 32) {
    int r = t >> 6, c = t & 63;
    int a = (c < 32) ? sSi[wave][r] : sSj[wave][r];
    hcw[a_idx(r, c)] = (_Float16)h[(size_t)a * 32 + (c & 31)];
  }
  __syncthreads();

  int n = lane & 15, mb = (lane & 16) ? 8 : 0;

  // GEMM1: [16x64]@[64x64pad] -> filter pre-activation
  {
    v16h a0f = frag_ld(&sHc[wave][0], lane);
    v16h a1f = frag_ld(&sHc[wave][32], lane);
#pragma unroll
    for (int nt = 0; nt < 4; ++nt) {
      v8f acc = {};
      acc = wmma32(a0f, frag_ld(&sWm[(0 * 4 + nt) * 32], lane), acc);
      acc = wmma32(a1f, frag_ld(&sWm[(1 * 4 + nt) * 32], lane), acc);
      float bb = sBm[nt * 16 + n];
#pragma unroll
      for (int r = 0; r < 8; ++r)
        sG1[wave][(mb + r) * 64 + nt * 16 + n] = (_Float16)(acc[r] + bb);
    }
  }
  __syncthreads();

  // assemble e_in = [h_cat | rbf(d)*filt | d | 0-pad] (16 x 128), fragment layout
  for (int t = lane; t < 16 * 128; t += 32) {
    int r = t >> 7, c = t & 127;
    float val;
    if (c < 64) val = (float)hcw[a_idx(r, c)];
    else if (c < 114) {
      int k = c - 64;
      float d = sDv[wave][r];
      float diff = d - (float)k * (5.0f / 49.0f);
      val = __expf(-10.f * diff * diff) * (float)sG1[wave][r * 64 + k];
    } else if (c == 114) val = sDv[wave][r];
    else val = 0.f;
    einw[a_idx(r, c)] = (_Float16)val;
  }
  __syncthreads();

  // GEMM2: [16x128]@[128x32] + silu
  {
    v16h af[4];
#pragma unroll
    for (int kk = 0; kk < 4; ++kk) af[kk] = frag_ld(&sEin[wave][kk * 32], lane);
#pragma unroll
    for (int nt = 0; nt < 2; ++nt) {
      v8f acc = {};
#pragma unroll
      for (int kk = 0; kk < 4; ++kk)
        acc = wmma32(af[kk], frag_ld(&sW1[(kk * 2 + nt) * 32], lane), acc);
      float bb = sB1[nt * 16 + n];
#pragma unroll
      for (int r = 0; r < 8; ++r)
        midw[a_slot(mb + r, nt * 16 + n)] = (_Float16)silu_f(acc[r] + bb);
    }
  }
  __syncthreads();

  // GEMM3: [16x32]@[32x32] -> h_ij_edge (fragment LDS f16 + global f32)
  {
    v16h af = frag_ld(&sMid[wave][0], lane);
#pragma unroll
    for (int nt = 0; nt < 2; ++nt) {
      v8f z = {};
      v8f acc = wmma32(af, frag_ld(&sW2[nt * 32], lane), z);
      float bb = sB2[nt * 16 + n];
#pragma unroll
      for (int r = 0; r < 8; ++r) {
        float zv = acc[r] + bb;
        edgw[a_slot(mb + r, nt * 16 + n)] = (_Float16)zv;
        ws_edge[(size_t)(p0 + mb + r) * 32 + nt * 16 + n] = zv;
      }
    }
  }
  __syncthreads();

  // GEMM4: attention logits + celu(alpha=2)
  {
    v16h af = frag_ld(&sEdg[wave][0], lane);
    v8f z = {};
    v8f acc = wmma32(af, frag_ld(&sWa[0], lane), z);
    if (n < 8) {
#pragma unroll
      for (int r = 0; r < 8; ++r) {
        float zv = acc[r] + sBa[n];
        float cel = (zv > 0.f) ? zv : 2.f * (__expf(0.5f * zv) - 1.f);
        ws_log[(size_t)(p0 + mb + r) * 8 + n] = (n < 7) ? cel : 0.f;
      }
    }
  }
}

// ---------------------------------------------------------------------------
__global__ void k_segmax(const int* __restrict__ pair, const float* __restrict__ ws_log,
                         unsigned* __restrict__ seg_max) {
  int idx = blockIdx.x * blockDim.x + threadIdx.x;
  int st = gridDim.x * blockDim.x;
  for (; idx < NP * 8; idx += st) {
    int hh = idx & 7;
    if (hh >= 7) continue;
    int p = idx >> 3;
    atomicMax(&seg_max[pair[p] * 8 + hh], f2ord(ws_log[idx]));
  }
}

__global__ void k_expsum(const int* __restrict__ pair, float* __restrict__ ws_log,
                         const unsigned* __restrict__ seg_max, float* __restrict__ seg_sum) {
  int idx = blockIdx.x * blockDim.x + threadIdx.x;
  int st = gridDim.x * blockDim.x;
  for (; idx < NP * 8; idx += st) {
    int hh = idx & 7;
    if (hh >= 7) continue;
    int p = idx >> 3;
    int i = pair[p];
    float e = __expf(ws_log[idx] - ord2f(seg_max[i * 8 + hh]));
    ws_log[idx] = e;
    atomicAdd(&seg_sum[i * 8 + hh], e);
  }
}

// ---------------------------------------------------------------------------
// Semantic features + the big [P,224]@[224,224] mix GEMM + spatial scatter-adds.
__global__ void __launch_bounds__(128) k_semantic(
    const int* __restrict__ pair, const float* __restrict__ w_xmix, const float* __restrict__ w_vmix,
    const float* __restrict__ ws_dir, const float* __restrict__ ws_edge,
    const float* __restrict__ ws_log, const float* __restrict__ seg_sum,
    float* __restrict__ hsem, float* __restrict__ comb, float* __restrict__ dvb) {
  __shared__ v16h sB[7 * 32];           // shared w_xmix N-chunk: 7 K-tiles x 1 N
  __shared__ v16h sSem[4][7 * 32];      // A: 16x224 per wave
  __shared__ float sEdge[4][16 * 32];
  __shared__ float sAtt[4][16 * 8];
  __shared__ float sDir[4][16 * 3];
  __shared__ float sVx[4][16];
  __shared__ int sI[4][16];

  int tid = threadIdx.x;
  int wave = tid >> 5, lane = tid & 31;
  int p0 = (blockIdx.x * 4 + wave) * 16;
  _Float16* bw   = (_Float16*)sB;
  _Float16* semw = (_Float16*)&sSem[wave][0];

  for (int t = lane; t < 16 * 32; t += 32)
    sEdge[wave][t] = ws_edge[(size_t)p0 * 32 + t];
  for (int t = lane; t < 16 * 8; t += 32) {
    int r = t >> 3, hh = t & 7;
    float a = 0.f;
    if (hh < 7) {
      int p = p0 + r;
      float ss = seg_sum[pair[p] * 8 + hh];
      a = (ss > 0.f) ? ws_log[(size_t)p * 8 + hh] / ss : 0.f;
    }
    sAtt[wave][t] = a;
  }
  if (lane < 16) {
    int p = p0 + lane;
    sI[wave][lane] = pair[p];
    sDir[wave][lane * 3 + 0] = ws_dir[p * 3 + 0];
    sDir[wave][lane * 3 + 1] = ws_dir[p * 3 + 1];
    sDir[wave][lane * 3 + 2] = ws_dir[p * 3 + 2];
    sVx[wave][lane] = 0.f;
  }
  __syncthreads();

  // h_ij_semantic (c = f*7 + h) + scatter-add into h_i_semantic
  for (int t = lane; t < 16 * 224; t += 32) {
    int r = t / 224, c = t - r * 224;
    int f = c / 7, hh = c - f * 7;
    float val = sEdge[wave][r * 32 + f] * sAtt[wave][r * 8 + hh];
    semw[a_idx(r, c)] = (_Float16)val;
    atomicAdd(&hsem[(size_t)sI[wave][r] * 224 + c], val);
  }
  __syncthreads();

  v16h af[7];
#pragma unroll
  for (int kk = 0; kk < 7; ++kk) af[kk] = frag_ld(&sSem[wave][kk * 32], lane);

  int n = lane & 15, mb = (lane & 16) ? 8 : 0;
  for (int nt = 0; nt < 14; ++nt) {           // block-uniform N-tile loop
    for (int t = tid; t < 224 * 16; t += 128) {
      int k = t >> 4, c = t & 15;
      bw[b_idx(k, c, 1)] = (_Float16)w_xmix[(size_t)k * 224 + nt * 16 + c];
    }
    if (tid == 0 && nt < 13)
      __builtin_prefetch(&w_xmix[(nt + 1) * 16], 0, 1);  // global_prefetch_b8
    __syncthreads();

    v8f acc = {};
#pragma unroll
    for (int kk = 0; kk < 7; ++kk)
      acc = wmma32(af[kk], frag_ld(&sB[kk * 32], lane), acc);

    int cidx = nt * 16 + n;
    float wv = w_vmix[cidx];
#pragma unroll
    for (int r = 0; r < 8; ++r) {
      int row = mb + r;
      float xm = tanhf(acc[r]);
      int i = sI[wave][row];
      size_t cb = ((size_t)i * 224 + cidx) * 3;
      atomicAdd(&comb[cb + 0], sDir[wave][row * 3 + 0] * xm);
      atomicAdd(&comb[cb + 1], sDir[wave][row * 3 + 1] * xm);
      atomicAdd(&comb[cb + 2], sDir[wave][row * 3 + 2] * xm);
      atomicAdd(&sVx[wave][row], wv * xm);
    }
    __syncthreads();
  }

  if (lane < 16) {
    int i = sI[wave][lane];
    float s = sVx[wave][lane];
    atomicAdd(&dvb[i * 3 + 0], sDir[wave][lane * 3 + 0] * s);
    atomicAdd(&dvb[i * 3 + 1], sDir[wave][lane * 3 + 1] * s);
    atomicAdd(&dvb[i * 3 + 2], sDir[wave][lane * 3 + 2] * s);
  }
}

// ---------------------------------------------------------------------------
// Per-atom MLPs (spatial / node / velocity) + integration. 16 atoms/wave.
__global__ void __launch_bounds__(64) k_node(
    const float* __restrict__ h, const float* __restrict__ x, const float* __restrict__ v,
    const float* __restrict__ w_n1, const float* __restrict__ b_n1,
    const float* __restrict__ w_n2, const float* __restrict__ b_n2,
    const float* __restrict__ w_p1, const float* __restrict__ b_p1,
    const float* __restrict__ w_p2, const float* __restrict__ b_p2,
    const float* __restrict__ w_v1, const float* __restrict__ b_v1,
    const float* __restrict__ w_v2,
    const float* __restrict__ counts, const float* __restrict__ hsem,
    const float* __restrict__ comb, const float* __restrict__ dvb,
    float* __restrict__ out_h, float* __restrict__ out_x, float* __restrict__ out_v) {
  __shared__ v16h sWp1[14 * 32];        // 7 K x 2 N
  __shared__ v16h sWp2[2 * 32];
  __shared__ v16h sWn1[18 * 32];        // 9 K x 2 N
  __shared__ v16h sWn2[2 * 32];
  __shared__ v16h sWv1[2 * 32];
  __shared__ float sWv2[32];
  __shared__ float sBp1[32], sBp2[32], sBn1[32], sBn2[32], sBv1[32];
  __shared__ v16h sBuf[2][9 * 32];      // A: 16x288 (norm_sq then node_in)
  __shared__ v16h sM1[2][32];           // A: 16x32
  __shared__ _Float16 sM2[2][16 * 32];  // linear (elementwise only)
  __shared__ float sDen[2][16];

  int tid = threadIdx.x;
  {
    _Float16* wp1 = (_Float16*)sWp1;
    _Float16* wn1 = (_Float16*)sWn1;
    _Float16* wp2 = (_Float16*)sWp2;
    _Float16* wn2 = (_Float16*)sWn2;
    _Float16* wv1 = (_Float16*)sWv1;
    for (int t = tid; t < 224 * 32; t += 64) {
      int k = t >> 5, c = t & 31;
      wp1[b_idx(k, c, 2)] = (_Float16)w_p1[t];
    }
    for (int t = tid; t < 288 * 32; t += 64) {
      int k = t >> 5, c = t & 31;
      wn1[b_idx(k, c, 2)] = (_Float16)w_n1[t];
    }
    for (int t = tid; t < 32 * 32; t += 64) {
      int k = t >> 5, c = t & 31;
      wp2[b_idx(k, c, 2)] = (_Float16)w_p2[t];
      wn2[b_idx(k, c, 2)] = (_Float16)w_n2[t];
      wv1[b_idx(k, c, 2)] = (_Float16)w_v1[t];
    }
  }
  if (tid < 32) {
    sWv2[tid] = w_v2[tid];
    sBp1[tid] = b_p1[tid]; sBp2[tid] = b_p2[tid];
    sBn1[tid] = b_n1[tid]; sBn2[tid] = b_n2[tid];
    sBv1[tid] = b_v1[tid];
  }
  __syncthreads();

  int wave = tid >> 5, lane = tid & 31;
  int tile = blockIdx.x * 2 + wave;
  if (tile >= NA / 16) return;
  int a0 = tile * 16;
  int n = lane & 15, mb = (lane & 16) ? 8 : 0;
  _Float16* bufw = (_Float16*)&sBuf[wave][0];
  _Float16* m1w  = (_Float16*)&sM1[wave][0];

  if (lane < 16) sDen[wave][lane] = fmaxf(counts[a0 + lane], 1.f);

  // norm_sq[16][224] from comb_mean (fragment layout)
  for (int t = lane; t < 16 * 224; t += 32) {
    int r = t / 224, c = t - r * 224;
    int i = a0 + r;
    float den = fmaxf(counts[i], 1.f);
    size_t cb = ((size_t)i * 224 + c) * 3;
    float mx = comb[cb + 0] / den, my = comb[cb + 1] / den, mz = comb[cb + 2] / den;
    bufw[a_idx(r, c)] = (_Float16)(mx * mx + my * my + mz * mz);
  }

  // spatial: silu(silu(nsq@Wp1+b)@Wp2+b) -> sM2 (linear)
  {
    v16h af[7];
#pragma unroll
    for (int kk = 0; kk < 7; ++kk) af[kk] = frag_ld(&sBuf[wave][kk * 32], lane);
#pragma unroll
    for (int nt = 0; nt < 2; ++nt) {
      v8f acc = {};
#pragma unroll
      for (int kk = 0; kk < 7; ++kk)
        acc = wmma32(af[kk], frag_ld(&sWp1[(kk * 2 + nt) * 32], lane), acc);
      float bb = sBp1[nt * 16 + n];
#pragma unroll
      for (int r = 0; r < 8; ++r)
        m1w[a_slot(mb + r, nt * 16 + n)] = (_Float16)silu_f(acc[r] + bb);
    }
  }
  {
    v16h af = frag_ld(&sM1[wave][0], lane);
#pragma unroll
    for (int nt = 0; nt < 2; ++nt) {
      v8f z = {};
      v8f acc = wmma32(af, frag_ld(&sWp2[nt * 32], lane), z);
      float bb = sBp2[nt * 16 + n];
#pragma unroll
      for (int r = 0; r < 8; ++r)
        sM2[wave][(mb + r) * 32 + nt * 16 + n] = (_Float16)silu_f(acc[r] + bb);
    }
  }

  // node_in = [h | h_i_semantic | h_i_spatial]  (16 x 288, fragment layout)
  for (int t = lane; t < 16 * 288; t += 32) {
    int r = t / 288, c = t - r * 288;
    int i = a0 + r;
    float val;
    if (c < 32) val = h[(size_t)i * 32 + c];
    else if (c < 256) val = hsem[(size_t)i * 224 + (c - 32)];
    else val = (float)sM2[wave][r * 32 + (c - 256)];
    bufw[a_idx(r, c)] = (_Float16)val;
  }

  // node update: h + silu(silu(node_in@Wn1+b)@Wn2+b)
  {
    v16h af[9];
#pragma unroll
    for (int kk = 0; kk < 9; ++kk) af[kk] = frag_ld(&sBuf[wave][kk * 32], lane);
#pragma unroll
    for (int nt = 0; nt < 2; ++nt) {
      v8f acc = {};
#pragma unroll
      for (int kk = 0; kk < 9; ++kk)
        acc = wmma32(af[kk], frag_ld(&sWn1[(kk * 2 + nt) * 32], lane), acc);
      float bb = sBn1[nt * 16 + n];
#pragma unroll
      for (int r = 0; r < 8; ++r)
        m1w[a_slot(mb + r, nt * 16 + n)] = (_Float16)silu_f(acc[r] + bb);
    }
  }
  {
    v16h af = frag_ld(&sM1[wave][0], lane);
#pragma unroll
    for (int nt = 0; nt < 2; ++nt) {
      v8f z = {};
      v8f acc = wmma32(af, frag_ld(&sWn2[nt * 32], lane), z);
      float bb = sBn2[nt * 16 + n];
#pragma unroll
      for (int r = 0; r < 8; ++r) {
        int i = a0 + mb + r, col = nt * 16 + n;
        out_h[(size_t)i * 32 + col] = h[(size_t)i * 32 + col] + silu_f(acc[r] + bb);
      }
    }
  }

  // velocity gate: 2*sigmoid(silu(h@Wv1+b)@Wv2)  (h = K-tile 0 of node_in)
  {
    v16h af = frag_ld(&sBuf[wave][0], lane);
#pragma unroll
    for (int nt = 0; nt < 2; ++nt) {
      v8f z = {};
      v8f acc = wmma32(af, frag_ld(&sWv1[nt * 32], lane), z);
      float bb = sBv1[nt * 16 + n];
#pragma unroll
      for (int r = 0; r < 8; ++r)
        sM2[wave][(mb + r) * 32 + nt * 16 + n] = (_Float16)silu_f(acc[r] + bb);
    }
  }
  if (lane < 16) {
    int i = a0 + lane;
    float s = 0.f;
#pragma unroll
    for (int f = 0; f < 32; ++f) s += (float)sM2[wave][lane * 32 + f] * sWv2[f];
    float scale = 2.f / (1.f + __expf(-s));
    float den = sDen[wave][lane];
#pragma unroll
    for (int d3 = 0; d3 < 3; ++d3) {
      float vv = scale * v[i * 3 + d3] + dvb[i * 3 + d3] / den;
      out_v[i * 3 + d3] = vv;
      out_x[i * 3 + d3] = x[i * 3 + d3] + vv;
    }
  }
}

// ---------------------------------------------------------------------------
extern "C" void kernel_launch(void* const* d_in, const int* in_sizes, int n_in,
                              void* d_out, int out_size, void* d_ws, size_t ws_size,
                              hipStream_t stream) {
  (void)in_sizes; (void)n_in; (void)out_size; (void)ws_size;
  const float* h        = (const float*)d_in[0];
  const float* x        = (const float*)d_in[1];
  const float* v        = (const float*)d_in[2];
  const int*   pair     = (const int*)d_in[3];
  const float* w_mlp_in = (const float*)d_in[4];
  const float* b_mlp_in = (const float*)d_in[5];
  const float* w_e1     = (const float*)d_in[6];
  const float* b_e1     = (const float*)d_in[7];
  const float* w_e2     = (const float*)d_in[8];
  const float* b_e2     = (const float*)d_in[9];
  const float* w_att    = (const float*)d_in[10];
  const float* b_att    = (const float*)d_in[11];
  const float* w_n1     = (const float*)d_in[12];
  const float* b_n1     = (const float*)d_in[13];
  const float* w_n2     = (const float*)d_in[14];
  const float* b_n2     = (const float*)d_in[15];
  const float* w_p1     = (const float*)d_in[16];
  const float* b_p1     = (const float*)d_in[17];
  const float* w_p2     = (const float*)d_in[18];
  const float* b_p2     = (const float*)d_in[19];
  const float* w_v1     = (const float*)d_in[20];
  const float* b_v1     = (const float*)d_in[21];
  const float* w_v2     = (const float*)d_in[22];
  const float* w_xmix   = (const float*)d_in[23];
  const float* w_vmix   = (const float*)d_in[24];

  float* ws      = (float*)d_ws;
  float* ws_d    = ws;                                   // P
  float* ws_dir  = ws_d + NP;                            // 3P
  float* ws_edge = ws_dir + (size_t)3 * NP;              // 32P
  float* ws_log  = ws_edge + (size_t)32 * NP;            // 8P
  unsigned* seg_max = (unsigned*)(ws_log + (size_t)8 * NP); // 8N
  float* seg_sum = (float*)(seg_max + (size_t)8 * NA);   // 8N
  float* counts  = seg_sum + (size_t)8 * NA;             // N
  float* hsem    = counts + NA;                          // 224N
  float* comb    = hsem + (size_t)224 * NA;              // 672N
  float* dvb     = comb + (size_t)672 * NA;              // 3N

  float* out_h = (float*)d_out;
  float* out_x = out_h + (size_t)32 * NA;
  float* out_v = out_x + (size_t)3 * NA;

  // zero the accumulator region (seg_max .. dvb): (8+8+1+224+672+3)*N words
  k_init<<<1024, 256, 0, stream>>>(seg_max, (long long)916 * NA);
  k_edge<<<NP / 64, 128, 0, stream>>>(h, x, pair, w_mlp_in, b_mlp_in, w_e1, b_e1,
                                      w_e2, b_e2, w_att, b_att,
                                      ws_d, ws_dir, ws_edge, ws_log, counts);
  k_segmax<<<2560, 256, 0, stream>>>(pair, ws_log, seg_max);
  k_expsum<<<2560, 256, 0, stream>>>(pair, ws_log, seg_max, seg_sum);
  k_semantic<<<NP / 64, 128, 0, stream>>>(pair, w_xmix, w_vmix, ws_dir, ws_edge,
                                          ws_log, seg_sum, hsem, comb, dvb);
  k_node<<<(NA / 16 + 1) / 2, 64, 0, stream>>>(h, x, v, w_n1, b_n1, w_n2, b_n2,
                                               w_p1, b_p1, w_p2, b_p2, w_v1, b_v1, w_v2,
                                               counts, hsem, comb, dvb, out_h, out_x, out_v);
}
